// DiscreteDenseKoopmanOperator_52484500357781
// MI455X (gfx1250) — matrix operand order repaired and checked by reading, not compiled, not run on hardware
//
#include <hip/hip_runtime.h>

typedef __attribute__((ext_vector_type(16))) __bf16 v16bf;
typedef __attribute__((ext_vector_type(8)))  __bf16 v8bf;
typedef __attribute__((ext_vector_type(8)))  float  v8f;

#define Dsz 512
#define Tsteps 256
#define ZSTRIDE 520   // +16B pad -> lanes land 4 banks apart (1040B row stride)
#define KSTRIDE 264   // +16B pad -> lanes land 4 banks apart (528B row stride)

union U { v16bf v; v8bf h[2]; };

// One-time prep: KT[n][k] = bf16(K[k][n]) so the WMMA B-operand (column n of
// K over a contiguous K-range) is a contiguous 32B segment per lane.
__global__ void koopman_kt_prep(const float* __restrict__ K,
                                __bf16* __restrict__ KT) {
    const int n = blockIdx.x;   // column of K == row of KT
    const int k = threadIdx.x;
    KT[n * Dsz + k] = (__bf16)K[k * Dsz + n];
}

// One rollout step: z_next = z_cur @ K for this wave's two 16x16 column
// tiles; f32 accumulate, f32 store to out[:, t, :], bf16 round into zw.
__device__ __forceinline__ void koop_step(
    const __bf16* __restrict__ zr,   // LDS: current z tile (read)
    __bf16* __restrict__ zw,         // LDS: next z tile (write, other buffer)
    const U (&breg0)[8], const U (&breg1)[8],
    const __bf16* __restrict__ kh0, const __bf16* __restrict__ kh1,
    float* __restrict__ out, size_t ob,
    int arow, int half, int l15, int colbase) {
    v8f c0 = {}, c1 = {};

    // k in [0,256): A from LDS, B from registers (loop-invariant, 128 VGPRs)
#pragma unroll
    for (int ki = 0; ki < 8; ++ki) {
        const int kk = ki * 32;
        U a;
        a.h[0] = *(const v8bf*)&zr[arow + kk + half * 16];
        a.h[1] = *(const v8bf*)&zr[arow + kk + half * 16 + 8];
        c0 = __builtin_amdgcn_wmma_f32_16x16x32_bf16(
                 false, a.v, false, breg0[ki].v, (short)0, c0, false, false);
        c1 = __builtin_amdgcn_wmma_f32_16x16x32_bf16(
                 false, a.v, false, breg1[ki].v, (short)0, c1, false, false);
    }
    // k in [256,512): A from LDS, B from LDS-resident kt_hi
#pragma unroll
    for (int ki = 0; ki < 8; ++ki) {
        const int kk = 256 + ki * 32;
        U a, bb0, bb1;
        a.h[0]   = *(const v8bf*)&zr[arow + kk + half * 16];
        a.h[1]   = *(const v8bf*)&zr[arow + kk + half * 16 + 8];
        bb0.h[0] = *(const v8bf*)&kh0[ki * 32];
        bb0.h[1] = *(const v8bf*)&kh0[ki * 32 + 8];
        bb1.h[0] = *(const v8bf*)&kh1[ki * 32];
        bb1.h[1] = *(const v8bf*)&kh1[ki * 32 + 8];
        c0 = __builtin_amdgcn_wmma_f32_16x16x32_bf16(
                 false, a.v, false, bb0.v, (short)0, c0, false, false);
        c1 = __builtin_amdgcn_wmma_f32_16x16x32_bf16(
                 false, a.v, false, bb1.v, (short)0, c1, false, false);
    }

    // f32 step result -> out[b0+M, t, colbase(+16)+N]
#pragma unroll
    for (int r = 0; r < 8; ++r) {
        out[ob + (size_t)r * Tsteps * Dsz]      = c0[r];
        out[ob + (size_t)r * Tsteps * Dsz + 16] = c1[r];
    }
    // bf16 z_{t+1} into the *other* buffer (no read/write hazard on zr)
#pragma unroll
    for (int r = 0; r < 8; ++r) {
        const int zrw = (r + 8 * half) * ZSTRIDE;
        zw[zrw + colbase + l15]      = (__bf16)c0[r];
        zw[zrw + colbase + 16 + l15] = (__bf16)c1[r];
    }
    __syncthreads();  // z_{t+1} visible; zr free for overwrite next step
}

// 16 workgroups (one 16-row batch tile each), 512 threads = 16 waves
// (4 waves/SIMD -> ~256 VGPR budget). Wave w owns columns [32w, 32w+32).
// LDS: double-buffered bf16 z tile (2 x 16.3 KB) + high-K half of KT for
// all columns (264 KB) -> ~297 KB of the 320 KB WGP LDS. Steady state has
// zero global/scratch loads and exactly one barrier per step.
__global__ __launch_bounds__(512) void koopman_rollout(
    const float* __restrict__ z0, const __bf16* __restrict__ KT,
    float* __restrict__ out) {
    __shared__ __bf16 zb0[16 * ZSTRIDE];
    __shared__ __bf16 zb1[16 * ZSTRIDE];
    __shared__ __bf16 kt_hi[Dsz * KSTRIDE];   // k in [256,512), all columns

    const int tid     = threadIdx.x;
    const int lane    = tid & 31;
    const int wave    = tid >> 5;        // 0..15
    const int b0      = blockIdx.x * 16; // batch-row tile base
    const int half    = lane >> 4;       // 0 or 1
    const int l15     = lane & 15;
    const int colbase = wave * 32;

    // ---- init z tile (bf16 of z0 rows), padded stride ----
    for (int i = tid; i < 16 * Dsz; i += 512) {
        const int m = i >> 9;
        const int d = i & (Dsz - 1);
        zb0[m * ZSTRIDE + d] = (__bf16)z0[(size_t)(b0 + m) * Dsz + d];
    }
    // ---- stage high-K half of KT into LDS (all 512 columns) ----
    for (int idx = tid; idx < Dsz * 32; idx += 512) {
        const int c = idx >> 5;           // column
        const int j = (idx & 31) * 8;     // k' offset within [0,256)
        *(v8bf*)&kt_hi[c * KSTRIDE + j] =
            *(const v8bf*)&KT[(size_t)c * Dsz + 256 + j];
    }
    __syncthreads();

    // ---- low-K half of B (k in [0,256)) for this wave's 2 column tiles:
    //      2 x 8 x 8 = 128 VGPRs, loaded once, loop-invariant ----
    U breg0[8], breg1[8];
    {
        const __bf16* ktl0 = KT + (size_t)(colbase + l15) * Dsz + half * 16;
        const __bf16* ktl1 = KT + (size_t)(colbase + 16 + l15) * Dsz + half * 16;
#pragma unroll
        for (int ki = 0; ki < 8; ++ki) {
            breg0[ki].h[0] = *(const v8bf*)&ktl0[ki * 32];
            breg0[ki].h[1] = *(const v8bf*)&ktl0[ki * 32 + 8];
            breg1[ki].h[0] = *(const v8bf*)&ktl1[ki * 32];
            breg1[ki].h[1] = *(const v8bf*)&ktl1[ki * 32 + 8];
        }
    }

    // ---- per-lane invariant addresses ----
    const int arow = l15 * ZSTRIDE;
    const __bf16* kh0 = &kt_hi[(colbase + l15) * KSTRIDE + half * 16];
    const __bf16* kh1 = kh0 + 16 * KSTRIDE;   // column tile +16
    const size_t out_base =
        (size_t)(b0 + 8 * half) * Tsteps * Dsz + colbase + l15;

    // t-loop manually unrolled by 2 so both LDS buffer bases stay constant.
    for (int t = 0; t < Tsteps; t += 2) {
        koop_step(zb0, zb1, breg0, breg1, kh0, kh1, out,
                  out_base + (size_t)t * Dsz, arow, half, l15, colbase);
        koop_step(zb1, zb0, breg0, breg1, kh0, kh1, out,
                  out_base + (size_t)(t + 1) * Dsz, arow, half, l15, colbase);
    }
}

extern "C" void kernel_launch(void* const* d_in, const int* in_sizes, int n_in,
                              void* d_out, int out_size, void* d_ws, size_t ws_size,
                              hipStream_t stream) {
    const float* z0  = (const float*)d_in[0];  // [256, 512] f32
    const float* K   = (const float*)d_in[1];  // [512, 512] f32
    float*       out = (float*)d_out;          // [256, 256, 512] f32
    __bf16*      KT  = (__bf16*)d_ws;          // 512*512*2 = 512 KB scratch

    koopman_kt_prep<<<dim3(Dsz), dim3(Dsz), 0, stream>>>(K, KT);
    koopman_rollout<<<dim3(256 / 16), dim3(512), 0, stream>>>(z0, KT, out);
}
